// LSTM_1D_11458972745748
// MI455X (gfx1250) — matrix operand order repaired and checked by reading, not compiled, not run on hardware
//
#include <hip/hip_runtime.h>
#include <hip/hip_bf16.h>

typedef __attribute__((ext_vector_type(16))) _Float16 v16h;
typedef __attribute__((ext_vector_type(8)))  float    v8f;

#define T_LEN   2048
#define BATCH   512
#define HID     64
#define BT      16      // batch tile per workgroup (WMMA N)
#define NWAVE   8
#define XCH     64      // x-chunk (timesteps staged in LDS)

// Native gfx1250 V_TANH_F32 (TRANS32 op: 1 instruction, branch-free,
// co-executes with the WMMA pipe). Builtin when available, else inline asm.
__device__ __forceinline__ float tanh_hw(float x) {
#if __has_builtin(__builtin_amdgcn_tanhf)
    return __builtin_amdgcn_tanhf(x);
#else
    float y;
    asm volatile("v_tanh_f32 %0, %1" : "=v"(y) : "v"(x));
    return y;
#endif
}

// sigmoid(x) = 0.5*tanh(0.5*x) + 0.5  ->  2 FMA + 1 v_tanh_f32, no branches
__device__ __forceinline__ float sigmoid_hw(float x) {
    return __builtin_fmaf(0.5f, tanh_hw(0.5f * x), 0.5f);
}

// gatesT = Wperm @ h^T  per layer, per step.
// Gate-row permutation: tile m (16 rows), row M -> original gate row
//   g = (M%4)*64 + (4*m + M/4)   (each tile interleaves i,f,g,o of 4 units)
// h^T staged in LDS in B-operand order: row (K/16)*16 + n holds the 16
// contiguous f16 K-values for batch column n -> one aligned v16h load/lane.
__global__ __launch_bounds__(256)
void lstm2_fused(const float* __restrict__ aim_p,
                 const float* __restrict__ Wih0, const float* __restrict__ Whh0,
                 const float* __restrict__ bih0, const float* __restrict__ bhh0,
                 const float* __restrict__ Wih1, const float* __restrict__ Whh1,
                 const float* __restrict__ bih1, const float* __restrict__ bhh1,
                 const float* __restrict__ Wlin, const float* __restrict__ blin,
                 float* __restrict__ out)
{
    __shared__ alignas(32) _Float16 BA[2][4 * 16 * 16];  // h0^T, K=64, ping-pong
    __shared__ alignas(32) _Float16 BB[2][8 * 16 * 16];  // [h0_cur ; h1_prev]^T, K=128
    __shared__ float xbuf[XCH][BT];
    __shared__ float pbuf[BT][NWAVE];

    const int tid = threadIdx.x;
    const int w   = tid >> 5;          // wave id 0..7
    const int l   = tid & 31;          // lane
    const int n   = l & 15;            // batch column within tile (N) / A-row (M)
    const int hh  = l >> 4;            // lane half (selects K sub-chunk)
    const int bBase = blockIdx.x * BT;

    // zero-init hidden-state staging buffers (h=c=0 at t=0)
    for (int i = tid; i < 4 * 16 * 16; i += 256) { BA[0][i] = (_Float16)0.f; BA[1][i] = (_Float16)0.f; }
    for (int i = tid; i < 8 * 16 * 16; i += 256) { BB[0][i] = (_Float16)0.f; BB[1][i] = (_Float16)0.f; }

    // ---- load gate-permuted, WMMA-striped weight A-operands into registers (once) ----
    // A f16 16x32 order used here (same as B): lane M = l%16; elem e -> K=(l/16)*16+e
    v16h aA[2][2];   // layer0 recurrent, tiles tt, K-chunks c (K=64)
    v16h aB[2][4];   // layer1, K=128 concat: c<2 -> W_ih1 (input h0), c>=2 -> W_hh1
    {
        const int M = n;
        #pragma unroll
        for (int tt = 0; tt < 2; ++tt) {
            const int m = 2 * w + tt;
            const int g = (M & 3) * 64 + (4 * m + (M >> 2));
            #pragma unroll
            for (int c = 0; c < 2; ++c) {
                v16h a;
                #pragma unroll
                for (int e = 0; e < 16; ++e) {
                    const int kk = c * 32 + hh * 16 + e;
                    a[e] = (_Float16)Whh0[g * HID + kk];
                }
                aA[tt][c] = a;
            }
            #pragma unroll
            for (int c = 0; c < 4; ++c) {
                const float* Wsrc = (c < 2) ? Wih1 : Whh1;
                const int cc = c & 1;
                v16h a;
                #pragma unroll
                for (int e = 0; e < 16; ++e) {
                    const int kk = cc * 32 + hh * 16 + e;
                    a[e] = (_Float16)Wsrc[g * HID + kk];
                }
                aB[tt][c] = a;
            }
        }
    }

    // per-lane constants: this lane owns hidden units j = 8w + 4tt + 2hh + pp
    float biasA[2][2][4], biasB[2][2][4], wxA[2][2][4], wl[2][2];
    #pragma unroll
    for (int tt = 0; tt < 2; ++tt)
        #pragma unroll
        for (int pp = 0; pp < 2; ++pp) {
            const int j = 8 * w + 4 * tt + 2 * hh + pp;
            wl[tt][pp] = Wlin[j];
            #pragma unroll
            for (int q = 0; q < 4; ++q) {
                const int g = q * 64 + j;
                biasA[tt][pp][q] = bih0[g] + bhh0[g];
                biasB[tt][pp][q] = bih1[g] + bhh1[g];
                wxA[tt][pp][q]   = Wih0[g];
            }
        }

    const float bl = blin[0];
    float cA[2][2] = {{0.f, 0.f}, {0.f, 0.f}};
    float cB[2][2] = {{0.f, 0.f}, {0.f, 0.f}};

    __syncthreads();

    for (int t = 0; t < T_LEN; ++t) {
        const int p = t & 1;

        if ((t & (XCH - 1)) == 0) {               // stage next 64 timesteps of x
            #pragma unroll
            for (int r = 0; r < 4; ++r) {
                const int tr = (tid >> 4) + r * 16;
                xbuf[tr][tid & 15] = aim_p[(size_t)(t + tr) * 1024 + bBase + (tid & 15)];
            }
            __syncthreads();
        }

        // ---------- layer 0: gates0^T = Wperm_hh0 @ h0_prev^T ----------
        const v16h b0 = *(const v16h*)&BA[p][((0 * 2 + hh) * 16 + n) * 16];
        const v16h b1 = *(const v16h*)&BA[p][((1 * 2 + hh) * 16 + n) * 16];
        v8f d[2];
        #pragma unroll
        for (int tt = 0; tt < 2; ++tt) {
            v8f acc = {};
            acc = __builtin_amdgcn_wmma_f32_16x16x32_f16(false, aA[tt][0], false, b0, (short)0, acc, false, false);
            acc = __builtin_amdgcn_wmma_f32_16x16x32_f16(false, aA[tt][1], false, b1, (short)0, acc, false, false);
            d[tt] = acc;
        }

        const float xv = xbuf[t & (XCH - 1)][n];
        #pragma unroll
        for (int tt = 0; tt < 2; ++tt)
            #pragma unroll
            for (int pp = 0; pp < 2; ++pp) {
                const int j = 8 * w + 4 * tt + 2 * hh + pp;
                const float gi = sigmoid_hw(d[tt][4 * pp + 0] + xv * wxA[tt][pp][0] + biasA[tt][pp][0]);
                const float gf = sigmoid_hw(d[tt][4 * pp + 1] + xv * wxA[tt][pp][1] + biasA[tt][pp][1]);
                const float gg = tanh_hw   (d[tt][4 * pp + 2] + xv * wxA[tt][pp][2] + biasA[tt][pp][2]);
                const float go = sigmoid_hw(d[tt][4 * pp + 3] + xv * wxA[tt][pp][3] + biasA[tt][pp][3]);
                const float c_ = gf * cA[tt][pp] + gi * gg;
                cA[tt][pp] = c_;
                const float h_ = go * tanh_hw(c_);
                const int row = (j >> 4) * 16 + n;
                const int e   = j & 15;
                BA[p ^ 1][row * 16 + e] = (_Float16)h_;   // next step's layer-0 input
                BB[p]    [row * 16 + e] = (_Float16)h_;   // this step's layer-1 input (K 0..63)
            }
        __syncthreads();

        // ---------- layer 1: gates1^T = [Wih1|Whh1]perm @ [h0_cur ; h1_prev]^T ----------
        const v16h e0 = *(const v16h*)&BB[p][((0 * 2 + hh) * 16 + n) * 16];
        const v16h e1 = *(const v16h*)&BB[p][((1 * 2 + hh) * 16 + n) * 16];
        const v16h e2 = *(const v16h*)&BB[p][((2 * 2 + hh) * 16 + n) * 16];
        const v16h e3 = *(const v16h*)&BB[p][((3 * 2 + hh) * 16 + n) * 16];
        #pragma unroll
        for (int tt = 0; tt < 2; ++tt) {
            v8f acc = {};
            acc = __builtin_amdgcn_wmma_f32_16x16x32_f16(false, aB[tt][0], false, e0, (short)0, acc, false, false);
            acc = __builtin_amdgcn_wmma_f32_16x16x32_f16(false, aB[tt][1], false, e1, (short)0, acc, false, false);
            acc = __builtin_amdgcn_wmma_f32_16x16x32_f16(false, aB[tt][2], false, e2, (short)0, acc, false, false);
            acc = __builtin_amdgcn_wmma_f32_16x16x32_f16(false, aB[tt][3], false, e3, (short)0, acc, false, false);
            d[tt] = acc;
        }

        float partial = 0.f;
        #pragma unroll
        for (int tt = 0; tt < 2; ++tt)
            #pragma unroll
            for (int pp = 0; pp < 2; ++pp) {
                const int j = 8 * w + 4 * tt + 2 * hh + pp;
                const float gi = sigmoid_hw(d[tt][4 * pp + 0] + biasB[tt][pp][0]);
                const float gf = sigmoid_hw(d[tt][4 * pp + 1] + biasB[tt][pp][1]);
                const float gg = tanh_hw   (d[tt][4 * pp + 2] + biasB[tt][pp][2]);
                const float go = sigmoid_hw(d[tt][4 * pp + 3] + biasB[tt][pp][3]);
                const float c_ = gf * cB[tt][pp] + gi * gg;
                cB[tt][pp] = c_;
                const float h_ = go * tanh_hw(c_);
                const int row = (4 + (j >> 4)) * 16 + n;        // K 64..127 half for next step
                BB[p ^ 1][row * 16 + (j & 15)] = (_Float16)h_;
                partial += tanh_hw(h_) * wl[tt][pp];            // output head partial
            }
        partial += __shfl_xor(partial, 16, 32);                 // fold lane halves (same b)
        if (l < 16) pbuf[n][w] = partial;
        __syncthreads();

        if (tid < 16) {
            float s = bl;
            #pragma unroll
            for (int ww = 0; ww < NWAVE; ++ww) s += pbuf[tid][ww];
            out[(size_t)t * BATCH + bBase + tid] = s;
        }
    }
}

extern "C" void kernel_launch(void* const* d_in, const int* in_sizes, int n_in,
                              void* d_out, int out_size, void* d_ws, size_t ws_size,
                              hipStream_t stream) {
    (void)in_sizes; (void)n_in; (void)out_size; (void)d_ws; (void)ws_size;
    const float* aim_p = (const float*)d_in[2];
    const float* Wih0  = (const float*)d_in[4];
    const float* Whh0  = (const float*)d_in[5];
    const float* bih0  = (const float*)d_in[6];
    const float* bhh0  = (const float*)d_in[7];
    const float* Wih1  = (const float*)d_in[8];
    const float* Whh1  = (const float*)d_in[9];
    const float* bih1  = (const float*)d_in[10];
    const float* bhh1  = (const float*)d_in[11];
    const float* Wlin  = (const float*)d_in[12];
    const float* blin  = (const float*)d_in[13];
    float* out = (float*)d_out;

    dim3 grid(BATCH / BT);   // 32 workgroups, one per 16-column batch slice
    dim3 block(256);         // 8 wave32
    lstm2_fused<<<grid, block, 0, stream>>>(aim_p, Wih0, Whh0, bih0, bhh0,
                                            Wih1, Whh1, bih1, bhh1,
                                            Wlin, blin, out);
}
